// RQKMeansPlus_16406775070843
// MI455X (gfx1250) — matrix operand order repaired and checked by reading, not compiled, not run on hardware
//
#include <hip/hip_runtime.h>
#include <hip/hip_bf16.h>
#include <math.h>

typedef __attribute__((ext_vector_type(8)))  _Float16 v8h;
typedef __attribute__((ext_vector_type(16))) _Float16 v16h;
typedef __attribute__((ext_vector_type(8)))  float    v8f;

#define BATCH 32768
#define IN_DIM 768
#define EMB 128
#define HID 256
#define KCB 2048
#define NLEV 4

// ---------------- fragment loaders (contiguous k order k0 + 16*g + s, wave32) ----------------
// A (16x32 f16, MxK): lane l holds row M=l&15; halves 0..15 = K k0+g*16..+15
// (g = l>>4). Two contiguous 16B loads.
__device__ __forceinline__ v16h load_a_frag(const _Float16* A, int ld, int row, int k0, int g) {
    const _Float16* p = A + (size_t)row * ld + k0 + g * 16;
    v8h lo = *(const v8h*)p;
    v8h hi = *(const v8h*)(p + 8);
    v16h r;
#pragma unroll
    for (int i = 0; i < 8; i++) { r[i] = lo[i]; r[8 + i] = hi[i]; }
    return r;
}
// B (32x16 f16, KxN) from B^T storage [N][K]: lane l holds col N=l&15,
// halves = K k0+g*16 .. k0+g*16+15. One contiguous 32B load.
__device__ __forceinline__ v16h load_b_frag(const _Float16* Bt, int ld, int col, int k0, int g) {
    return *(const v16h*)(Bt + (size_t)col * ld + k0 + g * 16);
}

__device__ __forceinline__ v8f wmma_f16(v16h a, v16h b, v8f c) {
    return __builtin_amdgcn_wmma_f32_16x16x32_f16(false, a, false, b, (short)0, c, false, false);
}

__device__ __forceinline__ float gelu_exact(float v) {
    return 0.5f * v * (1.0f + erff(v * 0.70710678118654752f));
}

// register-blocked GEMM core: one wave computes 16 rows x 64 cols; A fragment
// loaded once per K-chunk and reused across 4 column tiles (4 WMMA per chunk).
__device__ __forceinline__ void gemm_16x64(const _Float16* __restrict__ A, const _Float16* __restrict__ Bt,
                                           int K, int row0, int col0, int g, int lr, v8f acc[4]) {
    for (int k0 = 0; k0 < K; k0 += 32) {
        v16h a = load_a_frag(A, K, row0 + lr, k0, g);
#pragma unroll
        for (int t = 0; t < 4; t++) {
            v16h b = load_b_frag(Bt, K, col0 + t * 16 + lr, k0, g);
            acc[t] = wmma_f16(a, b, acc[t]);
        }
    }
}

// ---------------- conversion / prep kernels ----------------
__global__ void k_zero_accum(float* a) {
    if (threadIdx.x < 2) a[threadIdx.x] = 0.0f;
}

__global__ void k_cvt_f16(const float* __restrict__ in, _Float16* __restrict__ out, size_t n) {
    size_t i = (size_t)blockIdx.x * blockDim.x + threadIdx.x;
    size_t stride = (size_t)gridDim.x * blockDim.x;
    for (; i < n; i += stride) out[i] = (_Float16)in[i];
}

// in: f32 [K][N] row-major  ->  out: f16 [N][K] (transposed)
__global__ void k_transpose_cvt(const float* __restrict__ in, _Float16* __restrict__ out, int K, int N) {
    int idx = blockIdx.x * blockDim.x + threadIdx.x;
    int total = K * N;
    if (idx < total) {
        int n = idx / K, k = idx - n * K;
        out[(size_t)n * K + k] = (_Float16)in[(size_t)k * N + n];
    }
}

__global__ void k_cnorm(const float* __restrict__ cb, float* __restrict__ cn, int total) {
    int e = blockIdx.x * blockDim.x + threadIdx.x;
    if (e < total) {
        const float* p = cb + (size_t)e * EMB;
        float s = 0.f;
#pragma unroll 8
        for (int i = 0; i < EMB; i++) s += p[i] * p[i];
        cn[e] = s;
    }
}

// ---------------- encoder ----------------
// h = gelu(x @ W1 + b1) : M=B, N=256, K=768 ; wave tile 16x64
__global__ void k_enc1(const _Float16* __restrict__ xf, const _Float16* __restrict__ w1t,
                       const float* __restrict__ b1, _Float16* __restrict__ h) {
    int lane = threadIdx.x, g = lane >> 4, lr = lane & 15;
    int row0 = blockIdx.x * 16, col0 = blockIdx.y * 64;
    v8f acc[4] = {};
    gemm_16x64(xf, w1t, IN_DIM, row0, col0, g, lr, acc);
#pragma unroll
    for (int t = 0; t < 4; t++) {
        int col = col0 + t * 16 + lr;
        float bias = b1[col];
#pragma unroll
        for (int j = 0; j < 8; j++) {
            int r = row0 + g * 8 + j;
            float v = gelu_exact(acc[t][j] + bias);
            h[(size_t)r * HID + col] = (_Float16)v;
        }
    }
}

// z = h @ W2 + x @ SC + b2 + scb : N=128, K=256 then K=768 ; wave tile 16x64
__global__ void k_enc2(const _Float16* __restrict__ h, const _Float16* __restrict__ w2t,
                       const _Float16* __restrict__ xf, const _Float16* __restrict__ sct,
                       const float* __restrict__ b2, const float* __restrict__ scb,
                       float* __restrict__ z) {
    int lane = threadIdx.x, g = lane >> 4, lr = lane & 15;
    int row0 = blockIdx.x * 16, col0 = blockIdx.y * 64;
    v8f acc[4] = {};
    gemm_16x64(h, w2t, HID, row0, col0, g, lr, acc);
    gemm_16x64(xf, sct, IN_DIM, row0, col0, g, lr, acc);
#pragma unroll
    for (int t = 0; t < 4; t++) {
        int col = col0 + t * 16 + lr;
        float bias = b2[col] + scb[col];
#pragma unroll
        for (int j = 0; j < 8; j++) {
            int r = row0 + g * 8 + j;
            z[(size_t)r * EMB + col] = acc[t][j] + bias;
        }
    }
}

// ---------------- fused 4-level residual quantization ----------------
// one wave per 32-row tile (2 WMMA row tiles): codebook B fragments loaded once
// per 16-entry tile and reused by both row tiles -> 8 WMMA per 4 B-chunk loads.
// argmin_k(||c||^2 - 2 r.c) == argmin of the full squared distance.
__global__ void k_rq(const float* __restrict__ z, const float* __restrict__ cb_f32,
                     const _Float16* __restrict__ cb_f16, const float* __restrict__ cnorm,
                     float* __restrict__ qsum_f32, _Float16* __restrict__ qsum_f16,
                     float* __restrict__ out_codes, float* __restrict__ accum) {
    __shared__ float res[32][132];   // padded stride: avoid LDS bank conflicts
    __shared__ float qs[32][132];
    __shared__ int   sidx[32];
    int lane = threadIdx.x, g = lane >> 4, lr = lane & 15;
    int row0 = blockIdx.x * 32;

    for (int row = 0; row < 32; ++row)
        for (int d = lane; d < EMB; d += 32) {
            res[row][d] = z[(size_t)(row0 + row) * EMB + d];
            qs[row][d] = 0.0f;
        }
    __syncthreads();

    for (int level = 0; level < NLEV; ++level) {
        // residual -> f16 A fragments for both row tiles (K = 128 -> 4 chunks)
        v16h afr[2][4];
#pragma unroll
        for (int rt = 0; rt < 2; rt++) {
#pragma unroll
            for (int c = 0; c < 4; c++) {
                int k0 = c * 32;
                v16h a;
#pragma unroll
                for (int i = 0; i < 8; i++) a[i] = (_Float16)res[rt * 16 + lr][k0 + g * 16 + i];
#pragma unroll
                for (int i = 0; i < 8; i++) a[8 + i] = (_Float16)res[rt * 16 + lr][k0 + g * 16 + 8 + i];
                afr[rt][c] = a;
            }
        }
        const _Float16* cbl = cb_f16 + (size_t)level * KCB * EMB;
        const float* cnl = cnorm + level * KCB;
        float best[2][8]; int bidx[2][8];
#pragma unroll
        for (int rt = 0; rt < 2; rt++)
#pragma unroll
            for (int j = 0; j < 8; j++) { best[rt][j] = 3.4e38f; bidx[rt][j] = 0; }

        for (int t = 0; t < KCB / 16; t++) {
            int entry = t * 16 + lr;
            const _Float16* bp = cbl + (size_t)entry * EMB + g * 16;
            __builtin_prefetch(bp + 16 * EMB, 0, 1);   // next tile -> global_prefetch_b8
            v16h bfr[4];
#pragma unroll
            for (int c = 0; c < 4; c++) bfr[c] = *(const v16h*)(bp + c * 32);
            v8f acc0 = {}, acc1 = {};
#pragma unroll
            for (int c = 0; c < 4; c++) {
                acc0 = wmma_f16(afr[0][c], bfr[c], acc0);
                acc1 = wmma_f16(afr[1][c], bfr[c], acc1);
            }
            float cn = cnl[entry];
#pragma unroll
            for (int j = 0; j < 8; j++) {
                float s0 = fmaf(-2.0f, acc0[j], cn);
                if (s0 < best[0][j]) { best[0][j] = s0; bidx[0][j] = entry; }
                float s1 = fmaf(-2.0f, acc1[j], cn);
                if (s1 < best[1][j]) { best[1][j] = s1; bidx[1][j] = entry; }
            }
        }
        // argmin across the 16 lanes of each half (rows rt*16 + g*8 + j)
#pragma unroll
        for (int rt = 0; rt < 2; rt++)
#pragma unroll
            for (int j = 0; j < 8; j++) {
                for (int m = 1; m < 16; m <<= 1) {
                    float ob = __shfl_xor(best[rt][j], m, 32);
                    int   oi = __shfl_xor(bidx[rt][j], m, 32);
                    if (ob < best[rt][j] || (ob == best[rt][j] && oi < bidx[rt][j])) {
                        best[rt][j] = ob; bidx[rt][j] = oi;
                    }
                }
            }
        if (lr == 0) {
#pragma unroll
            for (int rt = 0; rt < 2; rt++)
#pragma unroll
                for (int j = 0; j < 8; j++) {
                    int row = rt * 16 + g * 8 + j;
                    sidx[row] = bidx[rt][j];
                    out_codes[(size_t)(row0 + row) * NLEV + level] = (float)bidx[rt][j];
                }
        }
        __syncthreads();
        // residual / qsum update from f32 codebook
        const float* cbf = cb_f32 + (size_t)level * KCB * EMB;
        for (int row = 0; row < 32; ++row) {
            const float* e = cbf + (size_t)sidx[row] * EMB;
            for (int d = lane; d < EMB; d += 32) {
                float v = e[d];
                res[row][d] -= v;
                qs[row][d] += v;
            }
        }
        __syncthreads();
    }
    // write qsum; residual == z - qsum -> commit loss partial
    float cl = 0.0f;
    for (int row = 0; row < 32; ++row)
        for (int d = lane; d < EMB; d += 32) {
            float q = qs[row][d];
            qsum_f32[(size_t)(row0 + row) * EMB + d] = q;
            qsum_f16[(size_t)(row0 + row) * EMB + d] = (_Float16)q;
            float r = res[row][d];
            cl += r * r;
        }
    for (int m = 1; m < 32; m <<= 1) cl += __shfl_xor(cl, m, 32);
    if (lane == 0) atomicAdd(&accum[1], cl);
}

// ---------------- decoder ----------------
// hdec = gelu(q @ dW1 + db1) : N=256, K=128 ; wave tile 16x64
__global__ void k_dec1(const _Float16* __restrict__ qf, const _Float16* __restrict__ dw1t,
                       const float* __restrict__ db1, _Float16* __restrict__ hdec) {
    int lane = threadIdx.x, g = lane >> 4, lr = lane & 15;
    int row0 = blockIdx.x * 16, col0 = blockIdx.y * 64;
    v8f acc[4] = {};
    gemm_16x64(qf, dw1t, EMB, row0, col0, g, lr, acc);
#pragma unroll
    for (int t = 0; t < 4; t++) {
        int col = col0 + t * 16 + lr;
        float bias = db1[col];
#pragma unroll
        for (int j = 0; j < 8; j++) {
            int r = row0 + g * 8 + j;
            float v = gelu_exact(acc[t][j] + bias);
            hdec[(size_t)r * HID + col] = (_Float16)v;
        }
    }
}

// x_hat = hdec @ dW2 + db2 : N=768, K=256 ; wave tile 16x64 ; fused recon loss
__global__ void k_dec2(const _Float16* __restrict__ hdec, const _Float16* __restrict__ dw2t,
                       const float* __restrict__ db2, const float* __restrict__ x,
                       float* __restrict__ xhat, float* __restrict__ accum) {
    int lane = threadIdx.x, g = lane >> 4, lr = lane & 15;
    int row0 = blockIdx.x * 16, col0 = blockIdx.y * 64;
    v8f acc[4] = {};
    gemm_16x64(hdec, dw2t, HID, row0, col0, g, lr, acc);
    float ls = 0.0f;
#pragma unroll
    for (int t = 0; t < 4; t++) {
        int col = col0 + t * 16 + lr;
        float bias = db2[col];
#pragma unroll
        for (int j = 0; j < 8; j++) {
            int r = row0 + g * 8 + j;
            float v = acc[t][j] + bias;
            xhat[(size_t)r * IN_DIM + col] = v;
            float d = v - x[(size_t)r * IN_DIM + col];
            ls += d * d;
        }
    }
    for (int m = 1; m < 32; m <<= 1) ls += __shfl_xor(ls, m, 32);
    if (lane == 0) atomicAdd(&accum[0], ls);
}

__global__ void k_finalize(const float* __restrict__ accum, float* __restrict__ out) {
    float recon = accum[0] * (1.0f / ((float)BATCH * (float)IN_DIM));
    float commit = accum[1] * (1.0f / ((float)BATCH * (float)EMB));
    out[0] = recon + 0.25f * commit;   // recon_loss_st == recon_loss numerically
    out[1] = recon;
    out[2] = commit;
}

// ---------------- host launcher ----------------
extern "C" void kernel_launch(void* const* d_in, const int* in_sizes, int n_in,
                              void* d_out, int out_size, void* d_ws, size_t ws_size,
                              hipStream_t stream) {
    const float* x      = (const float*)d_in[0];
    const float* enc_w1 = (const float*)d_in[1];
    const float* enc_b1 = (const float*)d_in[2];
    const float* enc_w2 = (const float*)d_in[3];
    const float* enc_b2 = (const float*)d_in[4];
    const float* sc_w   = (const float*)d_in[5];
    const float* sc_b   = (const float*)d_in[6];
    const float* dec_w1 = (const float*)d_in[7];
    const float* dec_b1 = (const float*)d_in[8];
    const float* dec_w2 = (const float*)d_in[9];
    const float* dec_b2 = (const float*)d_in[10];
    const float* cbooks = (const float*)d_in[11];

    float* out = (float*)d_out;
    float* out_codes = out + 3;                        // [B,4] as float
    float* out_xhat  = out + 3 + (size_t)BATCH * NLEV; // [B,768]

    char* ws = (char*)d_ws;
    size_t off = 0;
    auto alloc = [&](size_t bytes) { char* p = ws + off; off = (off + bytes + 255) & ~(size_t)255; return p; };
    _Float16* x_f16   = (_Float16*)alloc((size_t)BATCH * IN_DIM * 2);
    _Float16* h_f16   = (_Float16*)alloc((size_t)BATCH * HID * 2);
    float*    z_f32   = (float*)   alloc((size_t)BATCH * EMB * 4);
    float*    q_f32   = (float*)   alloc((size_t)BATCH * EMB * 4);
    _Float16* q_f16   = (_Float16*)alloc((size_t)BATCH * EMB * 2);
    _Float16* hd_f16  = (_Float16*)alloc((size_t)BATCH * HID * 2);
    _Float16* w1t     = (_Float16*)alloc((size_t)HID * IN_DIM * 2);
    _Float16* w2t     = (_Float16*)alloc((size_t)EMB * HID * 2);
    _Float16* sct     = (_Float16*)alloc((size_t)EMB * IN_DIM * 2);
    _Float16* dw1t    = (_Float16*)alloc((size_t)HID * EMB * 2);
    _Float16* dw2t    = (_Float16*)alloc((size_t)IN_DIM * HID * 2);
    _Float16* cb_f16  = (_Float16*)alloc((size_t)NLEV * KCB * EMB * 2);
    float*    cnorm   = (float*)   alloc((size_t)NLEV * KCB * 4);
    float*    accum   = (float*)   alloc(2 * 4);
    (void)ws_size; (void)in_sizes; (void)n_in; (void)out_size;

    k_zero_accum<<<1, 32, 0, stream>>>(accum);

    // conversions
    k_cvt_f16<<<4096, 256, 0, stream>>>(x, x_f16, (size_t)BATCH * IN_DIM);
    k_cvt_f16<<<1024, 256, 0, stream>>>(cbooks, cb_f16, (size_t)NLEV * KCB * EMB);
    k_transpose_cvt<<<(IN_DIM * HID + 255) / 256, 256, 0, stream>>>(enc_w1, w1t, IN_DIM, HID);
    k_transpose_cvt<<<(HID * EMB + 255) / 256, 256, 0, stream>>>(enc_w2, w2t, HID, EMB);
    k_transpose_cvt<<<(IN_DIM * EMB + 255) / 256, 256, 0, stream>>>(sc_w, sct, IN_DIM, EMB);
    k_transpose_cvt<<<(EMB * HID + 255) / 256, 256, 0, stream>>>(dec_w1, dw1t, EMB, HID);
    k_transpose_cvt<<<(HID * IN_DIM + 255) / 256, 256, 0, stream>>>(dec_w2, dw2t, HID, IN_DIM);
    k_cnorm<<<(NLEV * KCB + 255) / 256, 256, 0, stream>>>(cbooks, cnorm, NLEV * KCB);

    // encoder (wave tiles 16x64)
    k_enc1<<<dim3(BATCH / 16, HID / 64), 32, 0, stream>>>(x_f16, w1t, enc_b1, h_f16);
    k_enc2<<<dim3(BATCH / 16, EMB / 64), 32, 0, stream>>>(h_f16, w2t, x_f16, sct, enc_b2, sc_b, z_f32);

    // fused 4-level residual quantization (dominant WMMA workload, 32 rows/wave)
    k_rq<<<BATCH / 32, 32, 0, stream>>>(z_f32, cbooks, cb_f16, cnorm, q_f32, q_f16, out_codes, accum);

    // decoder + losses (wave tiles 16x64)
    k_dec1<<<dim3(BATCH / 16, HID / 64), 32, 0, stream>>>(q_f16, dw1t, dec_b1, hd_f16);
    k_dec2<<<dim3(BATCH / 16, IN_DIM / 64), 32, 0, stream>>>(hd_f16, dw2t, dec_b2, x, out_xhat, accum);

    k_finalize<<<1, 1, 0, stream>>>(accum, out);
}
